// UltimateRAv2_25383256719834
// MI455X (gfx1250) — matrix operand order repaired and checked by reading, not compiled, not run on hardware
//
#include <hip/hip_runtime.h>
#include <hip/hip_bf16.h>
#include <math.h>
#include <stdint.h>

// ---------------------------------------------------------------------------
// UltimateRAv2 on MI455X (gfx1250): bf16 WMMA GEMMs + LDS-staged flash attn.
// Shapes fixed by the reference: B=2, T=2048, C=1024, H=16, D=64, R=8.
// ---------------------------------------------------------------------------

typedef __bf16 bf16_t;
typedef __attribute__((ext_vector_type(16))) __bf16 bf16x16;
typedef __attribute__((ext_vector_type(8)))  __bf16 bf16x8;
typedef __attribute__((ext_vector_type(8)))  float  f32x8;
typedef int v4i_vec __attribute__((vector_size(16)));   // matches async builtin

#define BB    2
#define TT    2048
#define CC    1024
#define HH    16
#define DD    64
#define RR    8
#define DSTD  56
#define KQKV  1024          // reduction dim of qkv gemm
#define NQKV  3072          // q|k|v columns
#define MROWS 4096          // B*T
#define NEG_INF_F (-1.0e9f)

#ifndef __has_builtin
#define __has_builtin(x) 0
#endif
#if __has_builtin(__builtin_amdgcn_global_load_async_to_lds_b128) && \
    __has_builtin(__builtin_amdgcn_s_wait_asynccnt)
#define USE_ASYNC_LDS 1
#else
#define USE_ASYNC_LDS 0
#endif

// --------------------------- WMMA helpers ----------------------------------

__device__ __forceinline__ f32x8 wmma_bf16(bf16x16 a, bf16x16 b, f32x8 c) {
  // D = A(16x32 bf16) * B(32x16 bf16) + C(16x16 f32)
  return __builtin_amdgcn_wmma_f32_16x16x32_bf16(
      /*neg_a=*/false, a, /*neg_b=*/false, b,
      /*c_mod=*/(short)0, c, /*reuse_a=*/false, /*reuse_b=*/false);
}

// A fragment (16x32, MxK): lane&15 = row; lanes 0-15 hold K 0-7 / 16-23,
// lanes 16-31 hold K 8-15 / 24-31.  A is row-major with leading dim lda.
__device__ __forceinline__ bf16x16 load_a_frag(const bf16_t* A, int lda,
                                               int row0, int k0, int lane) {
  const int lr = lane & 15;
  const int hi = lane >> 4;
  const bf16_t* p = A + (size_t)(row0 + lr) * lda + k0 + hi * 8;
  bf16x8 lo = *(const bf16x8*)(p);
  bf16x8 hh = *(const bf16x8*)(p + 16);
  bf16x16 r;
#pragma unroll
  for (int i = 0; i < 8; ++i) { r[i] = lo[i]; r[8 + i] = hh[i]; }
  return r;
}

// B fragment (32x16, KxN) from a *transposed* (N-major) buffer Bt[n*ldb + k]:
// lanes 0-15 hold K 0-7 / 16-23 of column lane, lanes 16-31 hold K 8-15 / 24-31.
__device__ __forceinline__ bf16x16 load_b_frag(const bf16_t* Bt, int ldb,
                                               int n0, int k0, int lane) {
  const int lr = lane & 15;
  const int hi = lane >> 4;
  const bf16_t* p = Bt + (size_t)(n0 + lr) * ldb + k0 + hi * 8;
  bf16x8 lo = *(const bf16x8*)(p);
  bf16x8 hh = *(const bf16x8*)(p + 16);
  bf16x16 r;
#pragma unroll
  for (int i = 0; i < 8; ++i) { r[i] = lo[i]; r[8 + i] = hh[i]; }
  return r;
}

// B fragment from a 64x64 bf16 LDS tile (row stride 64 elements).
__device__ __forceinline__ bf16x16 load_b_frag_lds(const bf16_t* base,
                                                   int n0, int k0, int lane) {
  const int lr = lane & 15;
  const int hi = lane >> 4;
  const bf16_t* p = base + (n0 + lr) * 64 + k0 + hi * 8;
  bf16x8 lo = *(const bf16x8*)(p);
  bf16x8 hh = *(const bf16x8*)(p + 16);
  bf16x16 r;
#pragma unroll
  for (int i = 0; i < 8; ++i) { r[i] = lo[i]; r[8 + i] = hh[i]; }
  return r;
}

// Cooperative 64x64 bf16 tile stage: 256 threads x 2 chunks x 16B.
// gstride = global row stride (elements); LDS layout = row*64 + col.
__device__ __forceinline__ void stage_tile(const bf16_t* __restrict__ gsrc,
                                           size_t gstride, bf16_t* ldst, int tid) {
#pragma unroll
  for (int i = 0; i < 2; ++i) {
    const int c = tid + 256 * i;
    const int row = c >> 3;
    const int col = (c & 7) * 8;
    const bf16_t* src = gsrc + (size_t)row * gstride + col;
    bf16_t* dst = ldst + row * 64 + col;
#if USE_ASYNC_LDS
    __builtin_amdgcn_global_load_async_to_lds_b128(
        (__attribute__((address_space(1))) v4i_vec*)(uintptr_t)src,
        (__attribute__((address_space(3))) v4i_vec*)(uintptr_t)dst, 0, 0);
#else
    *(bf16x8*)dst = *(const bf16x8*)src;
#endif
  }
}

// --------------------------- prep kernels ----------------------------------

__global__ __launch_bounds__(256) void cvt_f32_bf16(const float* __restrict__ src,
                                                    bf16_t* __restrict__ dst, int n) {
  int i = blockIdx.x * 256 + threadIdx.x;
  if (i < n) dst[i] = (bf16_t)src[i];
}

// src is K x N row-major (f32); dst is N x K row-major (bf16) -> transposed.
__global__ __launch_bounds__(256) void cvt_transpose(const float* __restrict__ src,
                                                     bf16_t* __restrict__ dst,
                                                     int K, int N) {
  int idx = blockIdx.x * 256 + threadIdx.x;
  if (idx >= K * N) return;
  int k = idx / N, n = idx - k * N;
  dst[(size_t)n * K + k] = (bf16_t)src[idx];
}

// bias[h][j] = (d_bias[h][j] - mean_j(d_bias[h])) * w_disc[h]
__global__ __launch_bounds__(256) void bias_kernel(const float* __restrict__ d_bias,
                                                   const float* __restrict__ w_disc,
                                                   float* __restrict__ biasbuf) {
  __shared__ float red[256];
  const int h = blockIdx.x, tid = threadIdx.x;
  float s = 0.f;
  for (int j = tid; j < TT; j += 256) s += d_bias[h * TT + j];
  red[tid] = s;
  __syncthreads();
  for (int off = 128; off > 0; off >>= 1) {
    if (tid < off) red[tid] += red[tid + off];
    __syncthreads();
  }
  const float mean = red[0] * (1.0f / (float)TT);
  const float wd = w_disc[h];
  for (int j = tid; j < TT; j += 256)
    biasbuf[h * TT + j] = (d_bias[h * TT + j] - mean) * wd;
}

// --------------------------- GEMM 1: qkv -----------------------------------
// xb (4096 x 1024, bf16) * WaT (3072 x 1024, bf16, transposed-B layout).
// Epilogue scatters into q (B,H,T,D), k (B,H,T,D), v (B,H,D,T transposed).
__global__ __launch_bounds__(256) void gemm_qkv(const bf16_t* __restrict__ A,
                                                const bf16_t* __restrict__ Bt,
                                                bf16_t* __restrict__ qb,
                                                bf16_t* __restrict__ kb,
                                                bf16_t* __restrict__ vb) {
  const int lane = threadIdx.x & 31;
  const int w = threadIdx.x >> 5;               // 8 waves: 2 (M) x 4 (N)
  const int row0 = blockIdx.y * 128 + (w >> 2) * 64;
  const int col0 = blockIdx.x * 128 + (w & 3) * 32;

  f32x8 acc[4][2] = {};
  for (int k0 = 0; k0 < KQKV; k0 += 32) {
    bf16x16 af[4], bfr[2];
#pragma unroll
    for (int i = 0; i < 4; ++i) af[i] = load_a_frag(A, KQKV, row0 + 16 * i, k0, lane);
#pragma unroll
    for (int j = 0; j < 2; ++j) bfr[j] = load_b_frag(Bt, KQKV, col0 + 16 * j, k0, lane);
    {   // branch-free prefetch of next K-slab into near caches
      int kp = (k0 + 32 < KQKV) ? k0 + 32 : k0;
      __builtin_prefetch((const void*)(A + (size_t)(row0 + lane) * KQKV + kp), 0, 3);
      __builtin_prefetch((const void*)(Bt + (size_t)(col0 + lane) * KQKV + kp), 0, 3);
    }
#pragma unroll
    for (int i = 0; i < 4; ++i)
#pragma unroll
      for (int j = 0; j < 2; ++j) acc[i][j] = wmma_bf16(af[i], bfr[j], acc[i][j]);
  }

  const int lr = lane & 15;
  const int hi8 = (lane >> 4) * 8;
#pragma unroll
  for (int i = 0; i < 4; ++i)
#pragma unroll
    for (int j = 0; j < 2; ++j)
#pragma unroll
      for (int r = 0; r < 8; ++r) {
        int m = row0 + 16 * i + r + hi8;        // 0..4095 = b*T + t
        int n = col0 + 16 * j + lr;             // 0..3071
        int b = m >> 11, t = m & (TT - 1);
        int sec = n >> 10, c = n & (CC - 1);
        int h = c >> 6, d = c & 63;
        bf16_t val = (bf16_t)acc[i][j][r];
        size_t bh = (size_t)(b * HH + h);
        if (sec == 0)      qb[(bh * TT + t) * DD + d] = val;
        else if (sec == 1) kb[(bh * TT + t) * DD + d] = val;
        else               vb[(bh * DD + d) * TT + t] = val;   // transposed
      }
}

// --------------------- q_eff / k_eff augmentation (in place) ---------------
__global__ __launch_bounds__(256) void augment_kernel(bf16_t* __restrict__ qb,
                                                      bf16_t* __restrict__ kb,
                                                      const float* __restrict__ Wr,
                                                      const float* __restrict__ w_std,
                                                      const float* __restrict__ w_rec) {
  const int idx = blockIdx.x * 256 + threadIdx.x;   // row id in [0, B*H*T)
  const int h = (idx / TT) & (HH - 1);
  bf16_t* qr = qb + (size_t)idx * DD;
  bf16_t* kr = kb + (size_t)idx * DD;

  float qv[DD], kv[DD];
#pragma unroll
  for (int i = 0; i < 8; ++i) {
    bf16x8 a = *(const bf16x8*)(qr + i * 8);
    bf16x8 c = *(const bf16x8*)(kr + i * 8);
#pragma unroll
    for (int jj = 0; jj < 8; ++jj) { qv[i * 8 + jj] = (float)a[jj]; kv[i * 8 + jj] = (float)c[jj]; }
  }
  const float ws = sqrtf(fmaxf(w_std[h], 1e-8f));
  const float wraw = w_rec[h];
  const float wr = sqrtf(fmaxf(wraw, 1e-8f));
  const bool  ra = wraw > 0.1f;

  float QW[RR] = {}, KW[RR] = {};
#pragma unroll 4
  for (int dd = 0; dd < DD; ++dd) {
    float xq = qv[dd], xk = kv[dd];
#pragma unroll
    for (int rr = 0; rr < RR; ++rr) {
      float wv = Wr[dd * RR + rr];
      QW[rr] += xq * wv;
      KW[rr] += xk * wv;
    }
  }
  if (ra) {
#pragma unroll
    for (int dd = 0; dd < DSTD; ++dd) { qr[dd] = (bf16_t)(ws * qv[dd]); kr[dd] = (bf16_t)(ws * kv[dd]); }
#pragma unroll
    for (int rr = 0; rr < RR; ++rr) { qr[DSTD + rr] = (bf16_t)(wr * KW[rr]); kr[DSTD + rr] = (bf16_t)(wr * QW[rr]); }
  } else {
#pragma unroll
    for (int dd = 0; dd < DD; ++dd) { qr[dd] = (bf16_t)(ws * qv[dd]); kr[dd] = (bf16_t)(ws * kv[dd]); }
  }
}

// --------------------------- flash attention -------------------------------
// grid (T/128, H, B); 8 waves/block; each wave owns 16 queries.
// K/V 64-key tiles are staged into LDS cooperatively (async-to-LDS when the
// toolchain exposes the builtins) and shared by all 8 waves.
__global__ __launch_bounds__(256) void flash_kernel(const bf16_t* __restrict__ qe,
                                                    const bf16_t* __restrict__ ke,
                                                    const bf16_t* __restrict__ vt,
                                                    const float* __restrict__ bias,
                                                    bf16_t* __restrict__ Ob) {
  __shared__ __align__(16) bf16_t Ktile[64 * 64 + 32];   // keys x dims
  __shared__ __align__(16) bf16_t Vtile[64 * 64 + 32];   // dims x keys
  __shared__ __align__(16) bf16_t plds[8][16 * 16];      // per-wave P tile

  const int tid = threadIdx.x;
  const int lane = tid & 31;
  const int w = tid >> 5;
  const int lr = lane & 15;
  const int hi8 = (lane >> 4) * 8;

  const int qtile = blockIdx.x * 128;
  const int q0 = qtile + w * 16;
  const int h = blockIdx.y, b = blockIdx.z;
  const size_t bh = (size_t)b * HH + h;

  const bf16_t* qs = qe + bh * TT * DD;
  const bf16_t* ks = ke + bh * TT * DD;
  const bf16_t* vs = vt + bh * DD * TT;                  // (D, T) transposed
  const float* hb = bias + (size_t)h * TT;

  const bf16x16 qa0 = load_a_frag(qs, DD, q0, 0, lane);
  const bf16x16 qa1 = load_a_frag(qs, DD, q0, 32, lane);

  f32x8 acc[4] = {};
  float mrow[8], lrow[8];
#pragma unroll
  for (int r = 0; r < 8; ++r) { mrow[r] = -1e30f; lrow[r] = 0.f; }
  const float scale = 0.125f;                            // 1/sqrt(64)
  const int nkv = qtile + 128;                           // keys needed by block

  for (int jb64 = 0; jb64 < nkv; jb64 += 64) {
    __syncthreads();                                     // tiles free for reuse
    stage_tile(ks + (size_t)jb64 * DD, DD, Ktile, tid);  // 64 keys x 64 dims
    stage_tile(vs + jb64, TT, Vtile, tid);               // 64 dims x 64 keys
#if USE_ASYNC_LDS
    __builtin_amdgcn_s_wait_asynccnt(0);
#endif
    __syncthreads();                                     // tiles visible

#pragma unroll
    for (int sb = 0; sb < 4; ++sb) {
      const int jb = jb64 + sb * 16;
      if (jb > q0) continue;                             // causal: wave done here

      // ---- S tile = q_eff * k_eff^T (K = 64 -> 2 WMMAs), K frags from LDS
      bf16x16 kb0 = load_b_frag_lds(Ktile, sb * 16, 0, lane);
      bf16x16 kb1 = load_b_frag_lds(Ktile, sb * 16, 32, lane);
      f32x8 s = {};
      s = wmma_bf16(qa0, kb0, s);
      s = wmma_bf16(qa1, kb1, s);

      const float bj = hb[jb + lr];                      // key-dependent bias
      const bool diag = (jb == q0);
      float p[8];
#pragma unroll
      for (int r = 0; r < 8; ++r) {
        float x = s[r] * scale + bj;
        if (diag && lr > r + hi8) x = NEG_INF_F;         // mask col > row
        // row max over the 16 lanes of this half-wave
        float mx = x;
        mx = fmaxf(mx, __shfl_xor(mx, 1, 16));
        mx = fmaxf(mx, __shfl_xor(mx, 2, 16));
        mx = fmaxf(mx, __shfl_xor(mx, 4, 16));
        mx = fmaxf(mx, __shfl_xor(mx, 8, 16));
        const float mn = fmaxf(mrow[r], mx);
        const float sc = __expf(mrow[r] - mn);
        const float pv = __expf(x - mn);
        float ps = pv;
        ps += __shfl_xor(ps, 1, 16);
        ps += __shfl_xor(ps, 2, 16);
        ps += __shfl_xor(ps, 4, 16);
        ps += __shfl_xor(ps, 8, 16);
        lrow[r] = lrow[r] * sc + ps;
        mrow[r] = mn;
        acc[0][r] *= sc; acc[1][r] *= sc; acc[2][r] *= sc; acc[3][r] *= sc;
        p[r] = pv;
      }

      // ---- restage P (C layout) -> A layout through per-wave LDS tile
#pragma unroll
      for (int r = 0; r < 8; ++r)
        plds[w][(r + hi8) * 16 + lr] = (bf16_t)p[r];
      bf16x8 pl = *(const bf16x8*)&plds[w][lr * 16 + hi8];
      bf16x16 pa;
#pragma unroll
      for (int i = 0; i < 8; ++i) { pa[i] = pl[i]; pa[8 + i] = (bf16_t)0.0f; }

      // ---- O += P * V (4 N-tiles of 16 dims), V frags from LDS
#pragma unroll
      for (int t = 0; t < 4; ++t) {
        bf16x16 vbf = load_b_frag_lds(Vtile, t * 16, sb * 16, lane);
        acc[t] = wmma_bf16(pa, vbf, acc[t]);
      }
    }
  }

  // ---- normalize & write O in (B, T, C) bf16 layout for the proj GEMM
  bf16_t* orow = Ob + ((size_t)b * TT + q0) * CC + h * DD;
#pragma unroll
  for (int t = 0; t < 4; ++t)
#pragma unroll
    for (int r = 0; r < 8; ++r) {
      int row = r + hi8;
      int d = t * 16 + lr;
      orow[(size_t)row * CC + d] = (bf16_t)(acc[t][r] / lrow[r]);
    }
}

// --------------------------- GEMM 3: projection ----------------------------
// Ob (4096 x 1024 bf16) * WpT (1024 x 1024 bf16, transposed) -> f32 d_out.
__global__ __launch_bounds__(256) void gemm_proj(const bf16_t* __restrict__ A,
                                                 const bf16_t* __restrict__ Bt,
                                                 float* __restrict__ out) {
  const int lane = threadIdx.x & 31;
  const int w = threadIdx.x >> 5;
  const int row0 = blockIdx.y * 128 + (w >> 2) * 64;
  const int col0 = blockIdx.x * 128 + (w & 3) * 32;

  f32x8 acc[4][2] = {};
  for (int k0 = 0; k0 < CC; k0 += 32) {
    bf16x16 af[4], bfr[2];
#pragma unroll
    for (int i = 0; i < 4; ++i) af[i] = load_a_frag(A, CC, row0 + 16 * i, k0, lane);
#pragma unroll
    for (int j = 0; j < 2; ++j) bfr[j] = load_b_frag(Bt, CC, col0 + 16 * j, k0, lane);
    {
      int kp = (k0 + 32 < CC) ? k0 + 32 : k0;
      __builtin_prefetch((const void*)(A + (size_t)(row0 + lane) * CC + kp), 0, 3);
      __builtin_prefetch((const void*)(Bt + (size_t)(col0 + lane) * CC + kp), 0, 3);
    }
#pragma unroll
    for (int i = 0; i < 4; ++i)
#pragma unroll
      for (int j = 0; j < 2; ++j) acc[i][j] = wmma_bf16(af[i], bfr[j], acc[i][j]);
  }

  const int lr = lane & 15;
  const int hi8 = (lane >> 4) * 8;
#pragma unroll
  for (int i = 0; i < 4; ++i)
#pragma unroll
    for (int j = 0; j < 2; ++j)
#pragma unroll
      for (int r = 0; r < 8; ++r) {
        int m = row0 + 16 * i + r + hi8;
        int n = col0 + 16 * j + lr;
        out[(size_t)m * CC + n] = acc[i][j][r];
      }
}

// --------------------------- launcher --------------------------------------

extern "C" void kernel_launch(void* const* d_in, const int* in_sizes, int n_in,
                              void* d_out, int out_size, void* d_ws, size_t ws_size,
                              hipStream_t stream) {
  (void)in_sizes; (void)n_in; (void)out_size; (void)ws_size;
  const float* x      = (const float*)d_in[0];
  const float* W_attn = (const float*)d_in[1];
  const float* W_proj = (const float*)d_in[2];
  const float* W_rec  = (const float*)d_in[3];
  const float* w_std  = (const float*)d_in[4];
  const float* w_rec  = (const float*)d_in[5];
  const float* w_disc = (const float*)d_in[6];
  const float* d_bias = (const float*)d_in[7];
  float* out = (float*)d_out;

  char* ws = (char*)d_ws;
  size_t off = 0;
  auto wsalloc = [&](size_t bytes) -> void* {
    void* p = ws + off;
    off = (off + bytes + 255) & ~(size_t)255;
    return p;
  };
  bf16_t* xb   = (bf16_t*)wsalloc((size_t)MROWS * CC * 2);        //  8 MB
  bf16_t* WaT  = (bf16_t*)wsalloc((size_t)NQKV * CC * 2);         //  6 MB
  bf16_t* WpT  = (bf16_t*)wsalloc((size_t)CC * CC * 2);           //  2 MB
  bf16_t* qb   = (bf16_t*)wsalloc((size_t)BB * HH * TT * DD * 2); //  8 MB
  bf16_t* kb   = (bf16_t*)wsalloc((size_t)BB * HH * TT * DD * 2); //  8 MB
  bf16_t* vb   = (bf16_t*)wsalloc((size_t)BB * HH * TT * DD * 2 + 256); // +pad
  float*  bias = (float*) wsalloc((size_t)HH * TT * 4);
  bf16_t* Ob   = (bf16_t*)wsalloc((size_t)MROWS * CC * 2);        //  8 MB

  // prep: bf16 conversions (weights transposed into B-fragment-friendly layout)
  cvt_f32_bf16 <<<(MROWS * CC) / 256, 256, 0, stream>>>(x, xb, MROWS * CC);
  cvt_transpose<<<(CC * NQKV) / 256, 256, 0, stream>>>(W_attn, WaT, CC, NQKV);
  cvt_transpose<<<(CC * CC)   / 256, 256, 0, stream>>>(W_proj, WpT, CC, CC);
  bias_kernel  <<<HH, 256, 0, stream>>>(d_bias, w_disc, bias);

  // qkv GEMM (4096 x 3072 x 1024), epilogue scatters into heads (V transposed)
  gemm_qkv<<<dim3(NQKV / 128, MROWS / 128), 256, 0, stream>>>(xb, WaT, qb, kb, vb);

  // reciprocal augmentation: q_eff/k_eff in place
  augment_kernel<<<(BB * HH * TT) / 256, 256, 0, stream>>>(qb, kb, W_rec, w_std, w_rec);

  // flash attention with causal mask + key bias (LDS-staged K/V tiles)
  flash_kernel<<<dim3(TT / 128, HH, BB), 256, 0, stream>>>(qb, kb, vb, bias, Ob);

  // output projection -> f32 result
  gemm_proj<<<dim3(CC / 128, MROWS / 128), 256, 0, stream>>>(Ob, WpT, out);
}
